// EnhancedMetaWeightNetwork_86028194939511
// MI455X (gfx1250) — compile-verified, matrix operand order checked by reading, not run on hardware
//
#include <hip/hip_runtime.h>
#include <hip/hip_bf16.h>

// ---------------------------------------------------------------------------
// EnhancedMetaWeightNetwork for MI455X (gfx1250, wave32, WMMA + TDM).
// Big GEMMs: TENSOR_LOAD_TO_LDS double-buffered staging -> ds_load -> WMMA.
// Attention: flash-style, WMMA QK^T / PV with online softmax.
// ---------------------------------------------------------------------------

#define B_  4
#define S_  2048
#define H_  1024
#define NH_ 8
#define D_  128
#define BS_ (B_ * S_)        // 8192
#define H3_ (3 * H_)         // 3072
#define MD_ 256
#define MD2_ 128

typedef __attribute__((ext_vector_type(16))) __bf16 bf16x16;
typedef __attribute__((ext_vector_type(8)))  float  f32x8;
typedef __attribute__((ext_vector_type(4)))  unsigned int u32x4;
typedef __attribute__((ext_vector_type(8)))  int i32x8;
typedef __attribute__((ext_vector_type(4)))  int i32x4;

static __device__ inline f32x8 zero8() {
  f32x8 z;
#pragma unroll
  for (int i = 0; i < 8; ++i) z[i] = 0.0f;
  return z;
}

// WMMA wrapper: D = A(16x32 bf16) * B(32x16 bf16) + C (16x16 f32)
static __device__ inline f32x8 wmma_bf16(bf16x16 a, bf16x16 b, f32x8 c) {
  return __builtin_amdgcn_wmma_f32_16x16x32_bf16(
      false, a, false, b, (short)0, c, false, false);
}

// Load a 16x32 (rows x K) bf16 fragment from row-major storage (global or LDS)
// with leading dim `ld` (elements). Serves A-fragments and NT B-fragments.
// ISA layout: lane l holds row (l&15); halves 0..7 <- K=(l>>4)*8+0..7,
// halves 8..15 <- K=(l>>4)*8+16..23.
static __device__ inline bf16x16 load_frag_nt(const __bf16* base, int ld, int lane) {
  int r  = lane & 15;
  int ko = (lane >> 4) << 3;
  const __bf16* p = base + (size_t)r * ld + ko;
  bf16x16 f;
#pragma unroll
  for (int i = 0; i < 8; ++i) f[i] = p[i];
#pragma unroll
  for (int i = 0; i < 8; ++i) f[8 + i] = p[16 + i];
  return f;
}

// ---------------------------------------------------------------------------
// TDM: issue a 2D tile DMA (global -> LDS), bf16 elements.
// D# packing per CDNA5 ISA 8.3/8.4:
//   group0: [1:0]=count=1, [63:32]=lds_addr, [120:64]=global_addr, [127:126]=2
//   group1: [17:16]=data_size(1 => 2B), [79:48]=tensor_dim0, [111:80]=tensor_dim1,
//           [127:112]=tile_dim0, [143:128]=tile_dim1, [207:160]=tensor_dim0_stride
// ---------------------------------------------------------------------------
static __device__ inline void tdm_load_2d(unsigned lds_addr, const void* gptr,
                                          unsigned tile_k, unsigned tile_rows,
                                          unsigned row_stride_elems) {
  unsigned long long ga = (unsigned long long)(uintptr_t)gptr;
  u32x4 g0;
  g0[0] = 1u;                                       // count = 1 (valid D#)
  g0[1] = lds_addr;                                 // LDS byte address
  g0[2] = (unsigned)(ga & 0xFFFFFFFFu);             // global addr lo
  g0[3] = (unsigned)((ga >> 32) & 0x1FFFFFFu) | (2u << 30);  // addr hi | type=2
  unsigned d0 = tile_k, d1 = tile_rows;
  unsigned long long st0 = row_stride_elems;
  i32x8 g1;
  g1[0] = (int)(1u << 16);                          // data_size = 1 (2 bytes)
  g1[1] = (int)((d0 & 0xFFFFu) << 16);              // tensor_dim0 [15:0]
  g1[2] = (int)((d0 >> 16) | ((d1 & 0xFFFFu) << 16));   // dim0 hi | dim1 lo
  g1[3] = (int)((d1 >> 16) | (tile_k << 16));       // dim1 hi | tile_dim0
  g1[4] = (int)(tile_rows & 0xFFFFu);               // tile_dim1 (tile_dim2=0)
  g1[5] = (int)(st0 & 0xFFFFFFFFull);               // tensor_dim0_stride lo
  g1[6] = (int)((st0 >> 32) & 0xFFFFull);           // stride hi (dim1_stride=0)
  g1[7] = 0;
  i32x4 z4 = {0, 0, 0, 0};
#if __clang_major__ >= 23
  i32x8 z8 = {0, 0, 0, 0, 0, 0, 0, 0};
  __builtin_amdgcn_tensor_load_to_lds(g0, g1, z4, z4, z8, 0);
#else
  __builtin_amdgcn_tensor_load_to_lds(g0, g1, z4, z4, 0);
#endif
}

// ---------------------------------------------------------------------------
// x = hidden + pos_embed, cast to bf16
// ---------------------------------------------------------------------------
__global__ void make_x_kernel(const float* __restrict__ hidden,
                              const float* __restrict__ pos,
                              __bf16* __restrict__ xb, int n) {
  int i = blockIdx.x * blockDim.x + threadIdx.x;
  if (i >= n) return;
  int sh = i & (S_ * H_ - 1);            // pos is [1,S,H], broadcast over B
  xb[i] = (__bf16)(hidden[i] + pos[sh]);
}

__global__ void cvt_f32_bf16_kernel(const float* __restrict__ src,
                                    __bf16* __restrict__ dst, int n) {
  int i = blockIdx.x * blockDim.x + threadIdx.x;
  if (i < n) dst[i] = (__bf16)src[i];
}

// ---------------------------------------------------------------------------
// TDM-staged NT GEMM: block = 8 waves -> C stripe 16 x 512.
// Per K-step (32): wave0 DMAs A(16x32) + B(512x32) into LDS (double-buffered,
// pipelined with s_wait_tensorcnt 2); every wave runs 4 WMMAs from LDS.
// EPI: 0 = f32 out, 1 = bf16 out, 2 = QKV scatter (q/k bf16, v transposed).
// N must be a multiple of 512; K a multiple of 32.
// ---------------------------------------------------------------------------
template <int EPI>
__global__ __launch_bounds__(256) void gemm_tdm_kernel(
    const __bf16* __restrict__ A, int lda,
    const __bf16* __restrict__ Bw, int ldb,
    const float* __restrict__ bias,
    float* __restrict__ Cf, __bf16* __restrict__ Cb, int ldc,
    __bf16* __restrict__ q, __bf16* __restrict__ kk, __bf16* __restrict__ vT,
    int N, int K) {
  __shared__ __bf16 sA[2][16 * 32];
  __shared__ __bf16 sB[2][512 * 32];
  int lane = threadIdx.x & 31;
  int wv   = threadIdx.x >> 5;
  int tilesNb = N >> 9;
  int tM  = blockIdx.x / tilesNb;
  int tNb = blockIdx.x - tM * tilesNb;
  int row0 = tM << 4;
  int colB = tNb << 9;               // block column base
  int col0 = colB + (wv << 6);       // wave column base

  f32x8 acc[4];
#pragma unroll
  for (int j = 0; j < 4; ++j) acc[j] = zero8();

  int nsteps = K >> 5;
  if (wv == 0) {
    tdm_load_2d((unsigned)(uintptr_t)&sA[0][0], A + (size_t)row0 * lda, 32, 16, lda);
    tdm_load_2d((unsigned)(uintptr_t)&sB[0][0], Bw + (size_t)colB * ldb, 32, 512, ldb);
  }
  for (int step = 0; step < nsteps; ++step) {
    if (wv == 0) {
      if (step + 1 < nsteps) {
        int buf = (step + 1) & 1;
        int kn  = (step + 1) << 5;
        tdm_load_2d((unsigned)(uintptr_t)&sA[buf][0], A + (size_t)row0 * lda + kn,
                    32, 16, lda);
        tdm_load_2d((unsigned)(uintptr_t)&sB[buf][0], Bw + (size_t)colB * ldb + kn,
                    32, 512, ldb);
        __builtin_amdgcn_s_wait_tensorcnt(2);   // step's 2 DMAs complete
      } else {
        __builtin_amdgcn_s_wait_tensorcnt(0);
      }
    }
    __syncthreads();                            // publish LDS tiles
    int buf = step & 1;
    bf16x16 a = load_frag_nt(&sA[buf][0], 32, lane);
#pragma unroll
    for (int j = 0; j < 4; ++j) {
      bf16x16 b = load_frag_nt(&sB[buf][(wv * 64 + j * 16) * 32], 32, lane);
      acc[j] = wmma_bf16(a, b, acc[j]);
    }
    __syncthreads();                            // tiles consumed; safe to overwrite
  }

  int half = lane >> 4, cl = lane & 15;
#pragma unroll
  for (int j = 0; j < 4; ++j) {
#pragma unroll
    for (int v = 0; v < 8; ++v) {
      int r = row0 + v + 8 * half;
      int c = col0 + j * 16 + cl;
      float val = acc[j][v];
      if (bias) val += bias[c];
      if (EPI == 0) {
        Cf[(size_t)r * ldc + c] = val;
      } else if (EPI == 1) {
        Cb[(size_t)r * ldc + c] = (__bf16)val;
      } else {
        int sel = c >> 10;                     // 0=q 1=k 2=v
        int hd  = c & (H_ - 1);
        int hh  = hd >> 7;
        int d   = hd & (D_ - 1);
        int b   = r >> 11;
        int s   = r & (S_ - 1);
        size_t bh = (size_t)(b * NH_ + hh);
        if (sel == 0)      q [(bh * S_ + s) * D_ + d] = (__bf16)val;
        else if (sel == 1) kk[(bh * S_ + s) * D_ + d] = (__bf16)val;
        else               vT[(bh * D_ + d) * S_ + s] = (__bf16)val;
      }
    }
  }
}

// ---------------------------------------------------------------------------
// Generic NT GEMM (register path) for the small meta-MLP layers.
// Wave tile: 16 x 64.
// ---------------------------------------------------------------------------
__global__ __launch_bounds__(256) void gemm_nt_bf16_kernel(
    const __bf16* __restrict__ A, int lda,
    const __bf16* __restrict__ Bw, int ldb,
    const float* __restrict__ bias,
    float* __restrict__ Cf, __bf16* __restrict__ Cb, int ldc,
    int M, int N, int K, int accumulate) {
  int lane = threadIdx.x & 31;
  int wave = blockIdx.x * (blockDim.x >> 5) + (threadIdx.x >> 5);
  int tilesN = N >> 6;
  int tM = wave / tilesN;
  int tN = wave - tM * tilesN;
  if (tM >= (M >> 4)) return;
  int row0 = tM << 4, col0 = tN << 6;

  f32x8 acc[4];
#pragma unroll
  for (int j = 0; j < 4; ++j) acc[j] = zero8();

  for (int k = 0; k < K; k += 32) {
    bf16x16 a = load_frag_nt(A + (size_t)row0 * lda + k, lda, lane);
#pragma unroll
    for (int j = 0; j < 4; ++j) {
      bf16x16 b = load_frag_nt(Bw + (size_t)(col0 + j * 16) * ldb + k, ldb, lane);
      acc[j] = wmma_bf16(a, b, acc[j]);
    }
  }

  int half = lane >> 4, cl = lane & 15;
#pragma unroll
  for (int j = 0; j < 4; ++j) {
#pragma unroll
    for (int v = 0; v < 8; ++v) {
      int r = row0 + v + 8 * half;
      int c = col0 + j * 16 + cl;
      float val = acc[j][v];
      if (bias) val += bias[c];
      size_t idx = (size_t)r * ldc + c;
      if (accumulate) val += Cf[idx];
      if (Cb) Cb[idx] = (__bf16)val;
      else    Cf[idx] = val;
    }
  }
}

// ---------------------------------------------------------------------------
// Flash attention: one wave per (b, h, 16-query tile). Online softmax with
// half-wave shuffles; P re-laid out via LDS; PV against transposed V rows.
// ---------------------------------------------------------------------------
__global__ __launch_bounds__(128) void attention_kernel(
    const __bf16* __restrict__ q, const __bf16* __restrict__ k,
    const __bf16* __restrict__ vT, const int* __restrict__ amask,
    __bf16* __restrict__ ctx) {
  __shared__ __bf16 psh[4][16][32];
  int lane = threadIdx.x & 31;
  int w    = threadIdx.x >> 5;
  int bid  = blockIdx.x;
  int qg = bid & 31;
  int h  = (bid >> 5) & (NH_ - 1);
  int b  = bid >> 8;
  int qt = qg * 4 + w;                          // 16-query tile, 0..127

  const __bf16* qbase = q  + ((size_t)(b * NH_ + h) * S_ + qt * 16) * D_;
  const __bf16* kbase = k  + (size_t)(b * NH_ + h) * S_ * D_;
  const __bf16* vbase = vT + (size_t)(b * NH_ + h) * D_ * S_;

  bf16x16 qa[4];
#pragma unroll
  for (int i = 0; i < 4; ++i) qa[i] = load_frag_nt(qbase + i * 32, D_, lane);

  f32x8 O[8];
  float m_run[8], l_run[8];
#pragma unroll
  for (int t = 0; t < 8; ++t) O[t] = zero8();
#pragma unroll
  for (int v = 0; v < 8; ++v) { m_run[v] = -1e30f; l_run[v] = 0.0f; }

  const float scale = 0.08838834764831845f;     // 1/sqrt(128)
  int half = lane >> 4, cl = lane & 15;

  for (int kb = 0; kb < S_; kb += 32) {
    f32x8 s0 = zero8(), s1 = zero8();
#pragma unroll
    for (int i = 0; i < 4; ++i) {
      bf16x16 k0 = load_frag_nt(kbase + (size_t)kb * D_ + i * 32, D_, lane);
      s0 = wmma_bf16(qa[i], k0, s0);
      bf16x16 k1 = load_frag_nt(kbase + (size_t)(kb + 16) * D_ + i * 32, D_, lane);
      s1 = wmma_bf16(qa[i], k1, s1);
    }
    int ok0 = amask[b * S_ + kb + cl];
    int ok1 = amask[b * S_ + kb + 16 + cl];
#pragma unroll
    for (int v = 0; v < 8; ++v) {
      s0[v] = ok0 ? s0[v] * scale : -1e9f;
      s1[v] = ok1 ? s1[v] * scale : -1e9f;
    }
#pragma unroll
    for (int v = 0; v < 8; ++v) {
      float mv = fmaxf(s0[v], s1[v]);
#pragma unroll
      for (int off = 8; off >= 1; off >>= 1)
        mv = fmaxf(mv, __shfl_xor(mv, off, 32));
      float mn    = fmaxf(m_run[v], mv);
      float alpha = __expf(m_run[v] - mn);
      float p0    = __expf(s0[v] - mn);
      float p1    = __expf(s1[v] - mn);
      float rs = p0 + p1;
#pragma unroll
      for (int off = 8; off >= 1; off >>= 1)
        rs += __shfl_xor(rs, off, 32);
      l_run[v] = l_run[v] * alpha + rs;
      m_run[v] = mn;
#pragma unroll
      for (int t = 0; t < 8; ++t) O[t][v] *= alpha;
      int rrow = v + 8 * half;
      psh[w][rrow][cl]      = (__bf16)p0;
      psh[w][rrow][cl + 16] = (__bf16)p1;
    }
    __syncthreads();                            // publish P tile
    bf16x16 pa = load_frag_nt(&psh[w][0][0], 32, lane);
    __syncthreads();                            // protect LDS before next write
#pragma unroll
    for (int t = 0; t < 8; ++t) {
      bf16x16 vb = load_frag_nt(vbase + (size_t)(t * 16) * S_ + kb, S_, lane);
      O[t] = wmma_bf16(pa, vb, O[t]);
    }
  }

#pragma unroll
  for (int t = 0; t < 8; ++t) {
#pragma unroll
    for (int v = 0; v < 8; ++v) {
      int r = qt * 16 + v + 8 * half;
      int c = h * D_ + t * 16 + cl;
      float val = O[t][v] / l_run[v];
      ctx[((size_t)(b * S_ + r)) * H_ + c] = (__bf16)val;
    }
  }
}

// ---------------------------------------------------------------------------
// LayerNorm + ReLU over C columns (one block per row, blockDim == C)
// ---------------------------------------------------------------------------
__global__ void ln_relu_kernel(const float* __restrict__ y,
                               const float* __restrict__ g,
                               const float* __restrict__ be,
                               __bf16* outb, float* outf, int C) {
  __shared__ float red[256];
  int r = blockIdx.x, t = threadIdx.x;
  size_t idx = (size_t)r * C + t;
  float x = y[idx];
  red[t] = x;
  __syncthreads();
  for (int off = C >> 1; off > 0; off >>= 1) {
    if (t < off) red[t] += red[t + off];
    __syncthreads();
  }
  float mu = red[0] / C;
  __syncthreads();
  float d = x - mu;
  red[t] = d * d;
  __syncthreads();
  for (int off = C >> 1; off > 0; off >>= 1) {
    if (t < off) red[t] += red[t + off];
    __syncthreads();
  }
  float var = red[0] / C;
  float val = d * rsqrtf(var + 1e-5f) * g[t] + be[t];
  val = fmaxf(val, 0.0f);
  if (outb) outb[idx] = (__bf16)val;
  else      outf[idx] = val;
}

// ---------------------------------------------------------------------------
// Final: base = h2 . w3 + b3 ; weight = clip(base*(1+imp), .1, 5) ; mask
// ---------------------------------------------------------------------------
__global__ void final_kernel(const float* __restrict__ h2,
                             const float* __restrict__ w3,
                             const float* __restrict__ b3,
                             const int* __restrict__ tok,
                             const int* __restrict__ amask,
                             const float* __restrict__ table,
                             float* __restrict__ out) {
  int r = blockIdx.x * blockDim.x + threadIdx.x;
  if (r >= BS_) return;
  const float* hp = h2 + (size_t)r * MD2_;
  float acc = 0.0f;
#pragma unroll 8
  for (int i = 0; i < MD2_; ++i) acc += hp[i] * w3[i];
  acc += b3[0];
  float imp = table[tok[r]];
  float wgt = acc * (1.0f + imp);
  wgt = fminf(fmaxf(wgt, 0.1f), 5.0f);
  out[r] = amask[r] ? wgt : 0.0f;
}

// ---------------------------------------------------------------------------
extern "C" void kernel_launch(void* const* d_in, const int* in_sizes, int n_in,
                              void* d_out, int out_size, void* d_ws, size_t ws_size,
                              hipStream_t stream) {
  (void)in_sizes; (void)n_in; (void)out_size; (void)ws_size;
  const float* hidden    = (const float*)d_in[0];
  const int*   token_ids = (const int*)  d_in[1];
  const int*   amask     = (const int*)  d_in[2];
  const float* pos       = (const float*)d_in[3];
  const float* in_proj_w = (const float*)d_in[4];
  const float* in_proj_b = (const float*)d_in[5];
  const float* out_w     = (const float*)d_in[6];
  const float* out_b     = (const float*)d_in[7];
  const float* w1        = (const float*)d_in[8];
  const float* b1        = (const float*)d_in[9];
  const float* g1        = (const float*)d_in[10];
  const float* beta1     = (const float*)d_in[11];
  const float* w2        = (const float*)d_in[12];
  const float* b2        = (const float*)d_in[13];
  const float* g2        = (const float*)d_in[14];
  const float* beta2     = (const float*)d_in[15];
  const float* w3        = (const float*)d_in[16];
  const float* b3        = (const float*)d_in[17];
  const float* table     = (const float*)d_in[18];
  float* out = (float*)d_out;

  char* ws = (char*)d_ws;
  const size_t MB = 1ull << 20;
  __bf16* x_bf16   = (__bf16*)(ws + 0 * MB);     // 16 MB [BS,H]
  __bf16* att_bf16 = (__bf16*)(ws + 16 * MB);    // 16 MB [BS,H]
  __bf16* ctx_bf16 = (__bf16*)(ws + 32 * MB);    // 16 MB [BS,H]
  __bf16* q_bf16   = (__bf16*)(ws + 48 * MB);    // 16 MB [B,NH,S,D]
  __bf16* k_bf16   = (__bf16*)(ws + 64 * MB);    // 16 MB [B,NH,S,D]
  __bf16* vT_bf16  = (__bf16*)(ws + 80 * MB);    // 16 MB [B,NH,D,S]
  __bf16* Wqkv     = (__bf16*)(ws + 96 * MB);    //  6 MB
  __bf16* Wout     = (__bf16*)(ws + 102 * MB);   //  2 MB
  __bf16* W1       = (__bf16*)(ws + 104 * MB);   //  1 MB
  __bf16* W2       = (__bf16*)(ws + 105 * MB);   // 64 KB
  // aliases over q/k (free after attention):
  float*  y1_f32   = (float*) (ws + 48 * MB);    //  8 MB [BS,256]
  __bf16* h1_bf16  = (__bf16*)(ws + 56 * MB);    //  4 MB [BS,256]
  float*  y2_f32   = (float*) (ws + 64 * MB);    //  4 MB [BS,128]
  float*  h2_f32   = (float*) (ws + 68 * MB);    //  4 MB [BS,128]

  // 1) x = hidden + pos -> bf16
  make_x_kernel<<<(BS_ * H_) / 256, 256, 0, stream>>>(hidden, pos, x_bf16, BS_ * H_);

  // 2) weight conversions f32 -> bf16
  cvt_f32_bf16_kernel<<<(H3_ * H_) / 256, 256, 0, stream>>>(in_proj_w, Wqkv, H3_ * H_);
  cvt_f32_bf16_kernel<<<(H_ * H_) / 256, 256, 0, stream>>>(out_w, Wout, H_ * H_);
  cvt_f32_bf16_kernel<<<(MD_ * 2 * H_) / 256, 256, 0, stream>>>(w1, W1, MD_ * 2 * H_);
  cvt_f32_bf16_kernel<<<(MD2_ * MD_) / 256, 256, 0, stream>>>(w2, W2, MD2_ * MD_);

  // 3) QKV projection via TDM-staged GEMM + per-head scatter (v transposed)
  gemm_tdm_kernel<2><<<(BS_ / 16) * (H3_ / 512), 256, 0, stream>>>(
      x_bf16, H_, Wqkv, H_, in_proj_b,
      nullptr, nullptr, 0, q_bf16, k_bf16, vT_bf16, H3_, H_);

  // 4) flash attention -> ctx bf16 [BS,H]
  attention_kernel<<<B_ * NH_ * (S_ / 64), 128, 0, stream>>>(
      q_bf16, k_bf16, vT_bf16, amask, ctx_bf16);

  // 5) out projection via TDM-staged GEMM -> attended bf16
  gemm_tdm_kernel<1><<<(BS_ / 16) * (H_ / 512), 256, 0, stream>>>(
      ctx_bf16, H_, Wout, H_, out_b,
      nullptr, att_bf16, H_, nullptr, nullptr, nullptr, H_, H_);

  // 6) meta layer 1: y1 = [x, attended] @ w1^T + b1 (K=2048 split, 2 passes)
  {
    int waves = (BS_ / 16) * (MD_ / 64);         // 2048
    gemm_nt_bf16_kernel<<<waves / 8, 256, 0, stream>>>(
        x_bf16, H_, W1, 2 * H_, b1, y1_f32, nullptr, MD_,
        BS_, MD_, H_, 0);
    gemm_nt_bf16_kernel<<<waves / 8, 256, 0, stream>>>(
        att_bf16, H_, W1 + H_, 2 * H_, nullptr, y1_f32, nullptr, MD_,
        BS_, MD_, H_, 1);
  }

  // 7) LN + ReLU (C=256) -> h1 bf16
  ln_relu_kernel<<<BS_, MD_, 0, stream>>>(y1_f32, g1, beta1, h1_bf16, nullptr, MD_);

  // 8) meta layer 2: y2 = h1 @ w2^T + b2
  {
    int waves = (BS_ / 16) * (MD2_ / 64);        // 1024
    gemm_nt_bf16_kernel<<<waves / 8, 256, 0, stream>>>(
        h1_bf16, MD_, W2, MD_, b2, y2_f32, nullptr, MD2_,
        BS_, MD2_, MD_, 0);
  }

  // 9) LN + ReLU (C=128) -> h2 f32
  ln_relu_kernel<<<BS_, MD2_, 0, stream>>>(y2_f32, g2, beta2, nullptr, h2_f32, MD2_);

  // 10) final projection + importance gather + clamp + mask
  final_kernel<<<BS_ / 256, 256, 0, stream>>>(h2_f32, w3, b3, token_ids, amask,
                                              table, out);
}
